// BiLSTM_CRF_54528904790927
// MI455X (gfx1250) — matrix operand order repaired and checked by reading, not compiled, not run on hardware
//
#include <hip/hip_runtime.h>
#include <hip/hip_bf16.h>

// CRF forward (logZ) + Viterbi for S=1024, B=1024, T=32 on gfx1250.
//
// Key MI455X reasoning:
//  - Bandwidth floor: feats read 2x (fwd+vit) + pointers write ~= 384 MB
//    -> ~16us at 23.3 TB/s. The naive algorithm, however, needs S*B*T*T ~= 1G
//    v_exp_f32 transcendentals, which dominates. We rewrite logsumexp as
//    exp-space GEMM:  S[b,i] = sum_j exp(alpha[b,j]-m[b]) * exp(trans[i,j])
//    which is a 16x16x32 matmul per 16-batch tile -> V_WMMA_F32_16X16X32_F16
//    (codegen-confirmed builtin), cutting exp count 32x and moving the
//    reduction onto the matrix pipe. P in [0,1] and E <= e^5 fit f16; the
//    accumulator is f32.
//  - Batches are independent -> zero inter-wave synchronization. One kernel,
//    block-specialized: blocks [0,8) run forward (64 waves, 16 batches/wave,
//    alpha master copy in per-wave LDS), blocks [8,136) run Viterbi
//    (1024 waves, 1 batch/wave, lane=tag, readlane-broadcast max-plus with
//    first-match argmax). Both phases overlap on the machine.
//  - global_prefetch_b8 issued 4 steps ahead (per-step latency << HBM latency).
//  - ln(x) implemented as v_log_f32 * ln2 (1 trans + 1 mul) instead of the
//    precise libm expansion (~12 VALU): accuracy there is already bounded by
//    the f16 GEMM quantization, and log2(0) = -inf propagates identically.

typedef __attribute__((ext_vector_type(16))) _Float16 v16h;
typedef __attribute__((ext_vector_type(8)))  float    v8f;

#define S_LEN 1024
#define B_LEN 1024
#define T_LEN 32
#define START_TAG 29
#define STOP_TAG  30
#define NEG_INF   (-10000.0f)
#define FWD_BLOCKS 8
#define VIT_BLOCKS 128
#define PF_DIST 4

// group-of-32 swizzle patterns: offset = xor<<10 | and_mask(0x1f)
// pattern must be a frontend-constant -> template parameter.
template <int PAT>
__device__ __forceinline__ float xorf(float v) {
    return __int_as_float(__builtin_amdgcn_ds_swizzle(__float_as_int(v), PAT));
}
__device__ __forceinline__ float wave_max32(float v) {
    v = fmaxf(v, xorf<0x401F>(v)); // xor 16
    v = fmaxf(v, xorf<0x201F>(v)); // xor 8
    v = fmaxf(v, xorf<0x101F>(v)); // xor 4
    v = fmaxf(v, xorf<0x081F>(v)); // xor 2
    v = fmaxf(v, xorf<0x041F>(v)); // xor 1
    return v;
}
__device__ __forceinline__ float wave_sum32(float v) {
    v += xorf<0x401F>(v);
    v += xorf<0x201F>(v);
    v += xorf<0x101F>(v);
    v += xorf<0x081F>(v);
    v += xorf<0x041F>(v);
    return v;
}
__device__ __forceinline__ float rdlane(float v, int l) {
    return __int_as_float(__builtin_amdgcn_readlane(__float_as_int(v), l));
}
// raw v_log_f32 (base-2) * ln2 : 1 transcendental + 1 mul.
__device__ __forceinline__ float fast_ln(float x) {
    return __builtin_amdgcn_logf(x) * 0.69314718055994531f;
}

__global__ __launch_bounds__(256)
void BiLSTM_CRF_54528904790927_kernel(const float* __restrict__ feats,
                                      const float* __restrict__ mask,
                                      const float* __restrict__ trans,
                                      float* __restrict__ out,
                                      int*   __restrict__ ptrOut) {
    const int wave = threadIdx.x >> 5;
    const int lane = threadIdx.x & 31;

    // per-wave alpha tiles for the forward role: 8 waves x 16 batches x 32 tags
    __shared__ float aLds[8 * 16 * 32];

    if (blockIdx.x < FWD_BLOCKS) {
        // ---------------- Forward role: WMMA logsumexp scan ----------------
        float* aw = &aLds[wave * (16 * 32)];
        const int b0 = (blockIdx.x * 8 + wave) * 16;  // 16 batches per wave
        const int lm = lane & 15;
        const int hi = lane >> 4;

        // init alpha[b][j]
        #pragma unroll
        for (int bb = 0; bb < 16; ++bb)
            aw[bb * 32 + lane] = (lane == START_TAG) ? 0.0f : NEG_INF;

        // Build B operands: E[j, i] = exp(trans[i*T + j]) in f16 B-layout
        // (lanes 0-15: N=lane, K=0..7,16..23; lanes 16-31: N=lane-16, K=8..15,24..31)
        v16h e0, e1;
        {
            const int i0 = lm, i1 = lm + 16;

            #pragma unroll
            for (int e = 0; e < 16; ++e) {
                const int j = (e < 8 ? e : e + 8) + hi * 8;
                e0[e] = (_Float16)__expf(trans[i0 * T_LEN + j]);
                e1[e] = (_Float16)__expf(trans[i1 * T_LEN + j]);
            }
        }

        for (int s = 0; s < S_LEN; ++s) {
            if (s + PF_DIST < S_LEN) {
                // one 128B line per batch row; lanes 0-15 cover the 16 rows
                __builtin_prefetch(
                    &feats[((size_t)(s + PF_DIST) * B_LEN + b0 + lm) * T_LEN], 0, 0);
            }
            // ---- load alpha in f16-A layout (lane element e -> j) ----
            // j(e) = (e<8 ? e : e+8) + hi*8  : two runs of 8 consecutive floats
            float av[16];
            {
                const float* row = &aw[lm * 32 + hi * 8];
                const float4 r0 = *(const float4*)(row + 0);
                const float4 r1 = *(const float4*)(row + 4);
                const float4 r2 = *(const float4*)(row + 16);
                const float4 r3 = *(const float4*)(row + 20);
                av[0] = r0.x;  av[1] = r0.y;  av[2]  = r0.z;  av[3]  = r0.w;
                av[4] = r1.x;  av[5] = r1.y;  av[6]  = r1.z;  av[7]  = r1.w;
                av[8] = r2.x;  av[9] = r2.y;  av[10] = r2.z;  av[11] = r2.w;
                av[12] = r3.x; av[13] = r3.y; av[14] = r3.z;  av[15] = r3.w;
            }
            // ---- m[b] = max_j alpha[b,j] (16 local + cross-half swap) ----
            float mx = av[0];
            #pragma unroll
            for (int e = 1; e < 16; ++e) mx = fmaxf(mx, av[e]);
            const float m = fmaxf(mx, xorf<0x401F>(mx));  // SWAPX16

            // ---- A = exp(alpha - m) in f16 ----
            v16h a;
            #pragma unroll
            for (int e = 0; e < 16; ++e) a[e] = (_Float16)__expf(av[e] - m);

            // ---- D = A x E (+0) : two 16x16 tag halves ----
            v8f cz = {0.f, 0.f, 0.f, 0.f, 0.f, 0.f, 0.f, 0.f};
            v8f d0 = __builtin_amdgcn_wmma_f32_16x16x32_f16(
                false, a, false, e0, (short)0, cz, false, false);
            v8f d1 = __builtin_amdgcn_wmma_f32_16x16x32_f16(
                false, a, false, e1, (short)0, cz, false, false);

            // redistribute m to D layout (batch row = r + 8*hi)
            float msv[16];
            #pragma unroll
            for (int r = 0; r < 16; ++r) msv[r] = rdlane(m, r);

            #pragma unroll
            for (int r = 0; r < 8; ++r) {
                const int   bidx = b0 + r + hi * 8;
                const size_t base = ((size_t)s * B_LEN + bidx) * T_LEN;
                const float f0v = feats[base + lm];
                const float f1v = feats[base + 16 + lm];
                const float mkv = mask[(size_t)s * B_LEN + bidx];
                const float mv  = hi ? msv[r + 8] : msv[r];
                const float u0 = fast_ln(d0[r]) + mv + f0v;
                const float u1 = fast_ln(d1[r]) + mv + f1v;
                if (mkv != 0.0f) {  // masked step: alpha stays (EXEC-predicated store)
                    aw[(r + hi * 8) * 32 + lm]      = u0;
                    aw[(r + hi * 8) * 32 + 16 + lm] = u1;
                }
            }
        }

        // logZ[b] = logsumexp_j(alpha[b,j] + trans[STOP, j])
        const float tstop = trans[STOP_TAG * T_LEN + lane];
        #pragma unroll 4
        for (int bb = 0; bb < 16; ++bb) {
            const float v  = aw[bb * 32 + lane] + tstop;
            const float mx = wave_max32(v);
            const float sm = wave_sum32(__expf(v - mx));
            if (lane == 0) out[b0 + bb] = fast_ln(sm) + mx;
        }
    } else {
        // ---------------- Viterbi role: lane = tag, one batch per wave -----
        const int b = (blockIdx.x - FWD_BLOCKS) * 8 + wave;

        float trow[32];  // trans[i=lane][j]
        #pragma unroll
        for (int j = 0; j < 32; ++j) trow[j] = trans[lane * T_LEN + j];
        const float tstop = trans[STOP_TAG * T_LEN + lane];

        float sv = (lane == START_TAG) ? 0.0f : NEG_INF;

        for (int s = 0; s < S_LEN; ++s) {
            const size_t off = (size_t)s * B_LEN + b;
            const float f  = feats[off * T_LEN + lane];
            const float mk = mask[off];
            if (s + PF_DIST < S_LEN)
                __builtin_prefetch(&feats[(off + (size_t)PF_DIST * B_LEN) * T_LEN + lane], 0, 0);

            // best/argmax over j with first-match tie rule (strict >)
            float best = rdlane(sv, 0) + trow[0];
            int   bj   = 0;
            #pragma unroll
            for (int j = 1; j < 32; ++j) {
                const float c = rdlane(sv, j) + trow[j];
                if (c > best) { best = c; bj = j; }
            }
            ptrOut[off * T_LEN + lane] = bj;      // coalesced 128B per wave
            const float upd = best + f;
            sv = (mk != 0.0f) ? upd : sv;
        }

        const float v  = sv + tstop;
        const float mx = wave_max32(v);
        if (lane == 0) out[B_LEN + b] = mx;
    }
}

extern "C" void kernel_launch(void* const* d_in, const int* in_sizes, int n_in,
                              void* d_out, int out_size, void* d_ws, size_t ws_size,
                              hipStream_t stream) {
    (void)in_sizes; (void)n_in; (void)out_size; (void)d_ws; (void)ws_size;
    const float* feats = (const float*)d_in[0];
    const float* msk   = (const float*)d_in[1];
    const float* trans = (const float*)d_in[2];
    float* out = (float*)d_out;                 // [logZ(1024) | best(1024) | ptrs...]
    int*   ptr = (int*)d_out + 2 * B_LEN;       // pointers stored as raw int32

    dim3 grid(FWD_BLOCKS + VIT_BLOCKS);
    dim3 block(256);
    BiLSTM_CRF_54528904790927_kernel<<<grid, block, 0, stream>>>(feats, msk, trans, out, ptr);
}
